// AttnBlock_41223096107286
// MI455X (gfx1250) — matrix operand order repaired and checked by reading, not compiled, not run on hardware
//
#include <hip/hip_runtime.h>

typedef __bf16 bf16;
typedef __attribute__((ext_vector_type(16))) __bf16 v16bf;
typedef __attribute__((ext_vector_type(8)))  float  v8f;

#define B_   4
#define C_   512
#define HW_  4096
#define G_   32
#define CG_  16    // channels per group
#define EPS_ 1e-5f

// LDS row pitches chosen so row stride % 64 banks == 4 -> conflict-free b128 A-frag loads
#define QLDA 520   // 512 + 8  (1040 B rows)
#define PLDA 136   // 128 + 8  (272 B rows)

// ---------------------------------------------------------------------------
// CDNA5 wave32 WMMA fragment layouts (CDNA5 ISA §7.12.2)
// A (16x32 bf16) as loaded here: lane<16 -> row=lane, K=0..15 ; lane>=16 -> row=lane-16, K=16..31
// B (32x16 bf16), this code's K order (matches A, not AMD's table order): lane<16 -> col=lane, K=0..15 ; lane>=16 -> col=lane-16, K=16..31
// C/D (16x16 f32): elem i -> row = 8*(lane>>4)+i, col = lane&15
//
// B-side matrices are stored PRE-PACKED in fragment order:
//   tile (kb, nb) of 32(K) x 16(N), tile index = nb*(K/32)+kb  (K-tiles contiguous)
//   element within tile = lane*16 + slot, lane = ((k>>4)&1)*16 + (n&15), slot = k&15
// so a B fragment is one contiguous 32-byte read per lane (2x b128).
// ---------------------------------------------------------------------------
__device__ inline v16bf load_a_frag(const bf16* __restrict__ a, int lda) {
    int lane  = threadIdx.x & 31;
    int row   = lane & 15;
    int khalf = (lane >> 4) * 16;
    const bf16* p = a + (size_t)row * lda;
    v16bf f;
#pragma unroll
    for (int i = 0; i < 8; ++i) {
        f[i]     = p[khalf + i];        // K = khalf .. khalf+7
        f[i + 8] = p[khalf + 8 + i];    // K = khalf+8 .. khalf+15
    }
    return f;
}

__device__ inline v16bf load_b_frag_packed(const bf16* __restrict__ tile) {
    return *((const v16bf*)tile + (threadIdx.x & 31));
}

// packed element address for B[k][n] in a [Kdim][Ndim] matrix
__device__ inline size_t bpack_idx(int k, int n, int Kdim) {
    return ((size_t)(n >> 4) * (Kdim >> 5) + (k >> 5)) * 512 +
           ((((k >> 4) & 1) * 16 + (n & 15)) * 16) + (k & 15);
}

// ---------------------------------------------------------------------------
// GroupNorm statistics: one block per (batch, group)
// ---------------------------------------------------------------------------
__global__ __launch_bounds__(256) void gn_stats_kernel(
    const float* __restrict__ x, float* __restrict__ mean, float* __restrict__ rstd) {
    __shared__ float s_sum[256], s_sq[256];
    int bg = blockIdx.x;
    const float* xp = x + (size_t)bg * CG_ * HW_;   // group channels are contiguous
    float s = 0.f, ss = 0.f;
    for (int i = threadIdx.x; i < CG_ * HW_; i += 256) {
        float v = xp[i];
        s += v; ss += v * v;
    }
    s_sum[threadIdx.x] = s; s_sq[threadIdx.x] = ss;
    __syncthreads();
    for (int off = 128; off > 0; off >>= 1) {
        if (threadIdx.x < (unsigned)off) {
            s_sum[threadIdx.x] += s_sum[threadIdx.x + off];
            s_sq[threadIdx.x]  += s_sq[threadIdx.x + off];
        }
        __syncthreads();
    }
    if (threadIdx.x == 0) {
        const float inv_n = 1.0f / (CG_ * HW_);
        float m = s_sum[0] * inv_n;
        float var = s_sq[0] * inv_n - m * m;
        mean[bg] = m;
        rstd[bg] = rsqrtf(var + EPS_);
    }
}

// ---------------------------------------------------------------------------
// GroupNorm apply + f32 -> bf16, writing B-packed layout (K=channel, N=pixel)
// ---------------------------------------------------------------------------
__global__ __launch_bounds__(256) void gn_apply_kernel(
    const float* __restrict__ x, const float* __restrict__ gamma,
    const float* __restrict__ beta, const float* __restrict__ mean,
    const float* __restrict__ rstd, bf16* __restrict__ h) {
    size_t idx = (size_t)blockIdx.x * 256 + threadIdx.x;
    int n  = (int)(idx & (HW_ - 1));
    int c  = (int)((idx >> 12) & (C_ - 1));
    int b  = (int)(idx / ((size_t)C_ * HW_));
    int bg = (int)(idx / ((size_t)CG_ * HW_));
    float v = (x[idx] - mean[bg]) * rstd[bg] * gamma[c] + beta[c];
    h[(size_t)b * C_ * HW_ + bpack_idx(c, n, C_)] = (bf16)v;
}

__global__ __launch_bounds__(256) void cvt_bf16_kernel(
    const float* __restrict__ in, bf16* __restrict__ out, int n) {
    int i = blockIdx.x * 256 + threadIdx.x;
    if (i < n) out[i] = (bf16)in[i];
}

// ---------------------------------------------------------------------------
// WMMA GEMM:  D[M=512][N=HW] = A[512][512] * B_packed[512][HW] + bias
// Block: 8 waves stacked in M, each wave a 32x64 tile (2x4 frags, 8 WMMA/K-step)
// MODE 0: store bf16 B-packed [d][n], Kdim=C     (k matrix for S-matmul)
// MODE 1: store bf16 row-major [n][d]            (q^T, A-side of S-matmul)
// MODE 2: store f32 resid + D                    (final projection -> d_out)
// MODE 3: store bf16 B-packed [n][d], Kdim=HW    (v^T for O-matmul)
// ---------------------------------------------------------------------------
template <int MODE>
__global__ __launch_bounds__(256) void wmma_gemm_kernel(
    const bf16* __restrict__ A, const bf16* __restrict__ Bm,
    const float* __restrict__ bias, bf16* __restrict__ out_bf,
    const float* __restrict__ resid, float* __restrict__ out_f) {
    int wave = threadIdx.x >> 5;
    int lane = threadIdx.x & 31;
    int mbase = blockIdx.x * 256 + wave * 32;
    int nbase = blockIdx.y * 64;
    int b = blockIdx.z;
    const bf16* Bb = Bm + (size_t)b * C_ * HW_;

    v8f acc[2][4] = {{v8f{}, v8f{}, v8f{}, v8f{}}, {v8f{}, v8f{}, v8f{}, v8f{}}};
    for (int ks = 0; ks < C_; ks += 32) {
        v16bf af0 = load_a_frag(A + (size_t)mbase * C_ + ks, C_);
        v16bf af1 = load_a_frag(A + (size_t)(mbase + 16) * C_ + ks, C_);
#pragma unroll
        for (int t = 0; t < 4; ++t) {
            const bf16* tp = Bb + (((size_t)((nbase >> 4) + t) * (C_ >> 5) + (ks >> 5)) * 512);
            v16bf bfr = load_b_frag_packed(tp);
            acc[0][t] = __builtin_amdgcn_wmma_f32_16x16x32_bf16(
                false, af0, false, bfr, (short)0, acc[0][t], false, false);
            acc[1][t] = __builtin_amdgcn_wmma_f32_16x16x32_bf16(
                false, af1, false, bfr, (short)0, acc[1][t], false, false);
        }
    }
    int row0 = (lane >> 4) * 8;
    int col  = lane & 15;
#pragma unroll
    for (int mt = 0; mt < 2; ++mt) {
        int mb = mbase + mt * 16;
#pragma unroll
        for (int t = 0; t < 4; ++t) {
            int n = nbase + t * 16 + col;
#pragma unroll
            for (int i = 0; i < 8; ++i) {
                int d = mb + row0 + i;
                float v = acc[mt][t][i] + bias[d];
                if (MODE == 0) {
                    // d is K-dim: contiguous in i (slot = d&15)
                    bf16* dst = out_bf + (size_t)b * C_ * HW_ +
                                bpack_idx(mb + row0, n, C_);
                    dst[i] = (bf16)v;
                } else if (MODE == 1) {
                    out_bf[((size_t)b * HW_ + n) * C_ + d] = (bf16)v;
                } else if (MODE == 3) {
                    // n is K-dim, d is N-dim
                    out_bf[(size_t)b * HW_ * C_ + bpack_idx(n, d, HW_)] = (bf16)v;
                } else {
                    size_t o = ((size_t)b * C_ + d) * HW_ + n;
                    out_f[o] = resid[o] + v;
                }
            }
        }
    }
}

// ---------------------------------------------------------------------------
// Flash attention: block = 32 query rows (2 strips of 16), 8 waves.
// Per 128-key iteration: wave w computes S frags (one per strip) for keys
// [m0+16w, m0+16w+16) sharing each K B-fragment, online softmax across waves
// via LDS partials, then each wave accumulates its 64-wide channel slice of
// both strips' O (2 x 16x512) sharing each V^T B-fragment.
// K and V^T arrive pre-packed -> every B fragment is 2x global_load_b128.
// ---------------------------------------------------------------------------
__global__ __launch_bounds__(256) void flash_attn_kernel(
    const bf16* __restrict__ qT,   // [B][HW][C] row-major
    const bf16* __restrict__ Km,   // [B] B-packed (K=c, N=m)
    const bf16* __restrict__ vT,   // [B] B-packed (K=m, N=c)
    bf16* __restrict__ oT) {       // [B] B-packed (K=c, N=n)
    __shared__ bf16  q_lds[32 * QLDA];
    __shared__ bf16  p_lds[2 * 16 * PLDA];
    __shared__ float pmax[2][8][16];
    __shared__ float psum[2][8][16];

    int wave = threadIdx.x >> 5;
    int lane = threadIdx.x & 31;
    int b  = blockIdx.x / (HW_ / 32);
    int q0 = (blockIdx.x % (HW_ / 32)) * 32;

    // stage the 32x512 query strip into LDS (padded rows: 64 uint4 -> pitch 65)
    {
        const uint4* src = (const uint4*)(qT + ((size_t)b * HW_ + q0) * C_);
        uint4* dst = (uint4*)q_lds;
        for (int i = threadIdx.x; i < 32 * 64; i += 256) {
            int r = i >> 6, cc = i & 63;
            dst[r * (QLDA / 8) + cc] = src[i];
        }
    }
    __syncthreads();

    const bf16* Kb  = Km + (size_t)b * C_ * HW_;
    const bf16* vTb = vT + (size_t)b * HW_ * C_;

    const float scale = 0.044194173824159216f;  // 512^-0.5
    int row0  = (lane >> 4) * 8;
    int cbase = wave * 64;

    float m_st[2][8], l_st[2][8];
#pragma unroll
    for (int st = 0; st < 2; ++st)
#pragma unroll
        for (int i = 0; i < 8; ++i) { m_st[st][i] = -3.0e38f; l_st[st][i] = 0.f; }
    v8f o_acc[2][4] = {{v8f{}, v8f{}, v8f{}, v8f{}}, {v8f{}, v8f{}, v8f{}, v8f{}}};

    for (int m0 = 0; m0 < HW_; m0 += 128) {
        int mw = m0 + wave * 16;
        const bf16* Ktiles = Kb + (size_t)(mw >> 4) * (C_ >> 5) * 512;

        // prefetch next iteration's K tiles (global_prefetch_b8)
        if (m0 + 128 < HW_)
            __builtin_prefetch(Kb + (size_t)((mw + 128) >> 4) * (C_ >> 5) * 512, 0, 1);

        // ---- S = (q^T k) * scale : one 16x16 frag per strip, K-frag shared
        v8f s[2] = {v8f{}, v8f{}};
        for (int ks = 0; ks < C_; ks += 32) {
            v16bf bfr = load_b_frag_packed(Ktiles + (ks >> 5) * 512);
#pragma unroll
            for (int st = 0; st < 2; ++st) {
                v16bf af = load_a_frag(q_lds + st * 16 * QLDA + ks, QLDA);
                s[st] = __builtin_amdgcn_wmma_f32_16x16x32_bf16(
                    false, af, false, bfr, (short)0, s[st], false, false);
            }
        }

        // ---- per-row max over this wave's 16 key columns (within 16-lane half)
#pragma unroll
        for (int st = 0; st < 2; ++st) {
#pragma unroll
            for (int i = 0; i < 8; ++i) {
                float v = s[st][i] * scale;
                s[st][i] = v;
                v = fmaxf(v, __shfl_xor(v, 1));
                v = fmaxf(v, __shfl_xor(v, 2));
                v = fmaxf(v, __shfl_xor(v, 4));
                v = fmaxf(v, __shfl_xor(v, 8));
                if ((lane & 15) == 0) pmax[st][wave][row0 + i] = v;
            }
        }
        __syncthreads();

        // ---- new running max, P = exp(S - mnew) into LDS, partial row sums
        float mnew[2][8];
#pragma unroll
        for (int st = 0; st < 2; ++st) {
#pragma unroll
            for (int i = 0; i < 8; ++i) {
                float v = m_st[st][i];
#pragma unroll
                for (int w = 0; w < 8; ++w) v = fmaxf(v, pmax[st][w][row0 + i]);
                mnew[st][i] = v;
                float p = __expf(s[st][i] - v);
                p_lds[(st * 16 + row0 + i) * PLDA + wave * 16 + (lane & 15)] = (bf16)p;
                float r = p;
                r += __shfl_xor(r, 1);
                r += __shfl_xor(r, 2);
                r += __shfl_xor(r, 4);
                r += __shfl_xor(r, 8);
                if ((lane & 15) == 0) psum[st][wave][row0 + i] = r;
            }
        }
        __syncthreads();

        // ---- online-softmax state update + O rescale
#pragma unroll
        for (int st = 0; st < 2; ++st) {
#pragma unroll
            for (int i = 0; i < 8; ++i) {
                float tot = 0.f;
#pragma unroll
                for (int w = 0; w < 8; ++w) tot += psum[st][w][row0 + i];
                float alpha = __expf(m_st[st][i] - mnew[st][i]);
                l_st[st][i] = l_st[st][i] * alpha + tot;
                m_st[st][i] = mnew[st][i];
#pragma unroll
                for (int t = 0; t < 4; ++t) o_acc[st][t][i] *= alpha;
            }
        }

        // ---- O += P * V^T  (this wave's 64 channel columns, V-frag shared)
#pragma unroll
        for (int kk = 0; kk < 128; kk += 32) {
            v16bf afr[2];
#pragma unroll
            for (int st = 0; st < 2; ++st)
                afr[st] = load_a_frag(p_lds + st * 16 * PLDA + kk, PLDA);
#pragma unroll
            for (int t = 0; t < 4; ++t) {
                const bf16* tp = vTb +
                    ((size_t)((cbase + t * 16) >> 4) * (HW_ >> 5) + ((m0 + kk) >> 5)) * 512;
                v16bf bfr = load_b_frag_packed(tp);
#pragma unroll
                for (int st = 0; st < 2; ++st)
                    o_acc[st][t] = __builtin_amdgcn_wmma_f32_16x16x32_bf16(
                        false, afr[st], false, bfr, (short)0, o_acc[st][t], false, false);
            }
        }
        __syncthreads();  // protect pmax/psum/p_lds for next iteration
    }

    // ---- write O, B-packed (K=c, N=query n) for the final projection
    bf16* oTb = oT + (size_t)b * C_ * HW_;
#pragma unroll
    for (int st = 0; st < 2; ++st) {
#pragma unroll
        for (int t = 0; t < 4; ++t) {
            int c = cbase + t * 16 + (lane & 15);
#pragma unroll
            for (int i = 0; i < 8; ++i) {
                int n = q0 + st * 16 + row0 + i;
                oTb[bpack_idx(c, n, C_)] = (bf16)(o_acc[st][t][i] / l_st[st][i]);
            }
        }
    }
}

// ---------------------------------------------------------------------------
extern "C" void kernel_launch(void* const* d_in, const int* in_sizes, int n_in,
                              void* d_out, int out_size, void* d_ws, size_t ws_size,
                              hipStream_t stream) {
    (void)in_sizes; (void)n_in; (void)out_size; (void)ws_size;
    const float* x     = (const float*)d_in[0];
    const float* gamma = (const float*)d_in[1];
    const float* beta  = (const float*)d_in[2];
    const float* wq    = (const float*)d_in[3];
    const float* bq    = (const float*)d_in[4];
    const float* wk    = (const float*)d_in[5];
    const float* bk    = (const float*)d_in[6];
    const float* wv    = (const float*)d_in[7];
    const float* bv    = (const float*)d_in[8];
    const float* wo    = (const float*)d_in[9];
    const float* bo    = (const float*)d_in[10];
    float* out = (float*)d_out;

    char* ws = (char*)d_ws;
    auto take = [&](size_t bytes) -> void* {
        void* p = (void*)ws;
        ws += (bytes + 255) & ~(size_t)255;
        return p;
    };
    const size_t nW = (size_t)C_ * C_;          // 262144
    const size_t nT = (size_t)B_ * C_ * HW_;    // 8388608
    bf16* wq_bf = (bf16*)take(nW * 2);
    bf16* wk_bf = (bf16*)take(nW * 2);
    bf16* wv_bf = (bf16*)take(nW * 2);
    bf16* wo_bf = (bf16*)take(nW * 2);
    float* mean = (float*)take((size_t)B_ * G_ * 4);
    float* rstd = (float*)take((size_t)B_ * G_ * 4);
    bf16* h_bf  = (bf16*)take(nT * 2);   // B-packed (K=c, N=n)
    bf16* q_t   = (bf16*)take(nT * 2);   // row-major [B][HW][C]
    bf16* k_n   = (bf16*)take(nT * 2);   // B-packed (K=c, N=m)
    bf16* v_t   = (bf16*)take(nT * 2);   // B-packed (K=m, N=c)
    bf16* o_t   = (bf16*)take(nT * 2);   // B-packed (K=c, N=n)

    // weights -> bf16
    int wblk = (int)((nW + 255) / 256);
    cvt_bf16_kernel<<<wblk, 256, 0, stream>>>(wq, wq_bf, (int)nW);
    cvt_bf16_kernel<<<wblk, 256, 0, stream>>>(wk, wk_bf, (int)nW);
    cvt_bf16_kernel<<<wblk, 256, 0, stream>>>(wv, wv_bf, (int)nW);
    cvt_bf16_kernel<<<wblk, 256, 0, stream>>>(wo, wo_bf, (int)nW);

    // groupnorm
    gn_stats_kernel<<<B_ * G_, 256, 0, stream>>>(x, mean, rstd);
    gn_apply_kernel<<<(int)(nT / 256), 256, 0, stream>>>(x, gamma, beta, mean, rstd, h_bf);

    // q / k / v projections
    dim3 gg(C_ / 256, HW_ / 64, B_);
    wmma_gemm_kernel<1><<<gg, 256, 0, stream>>>(wq_bf, h_bf, bq, q_t, nullptr, nullptr);
    wmma_gemm_kernel<0><<<gg, 256, 0, stream>>>(wk_bf, h_bf, bk, k_n, nullptr, nullptr);
    wmma_gemm_kernel<3><<<gg, 256, 0, stream>>>(wv_bf, h_bf, bv, v_t, nullptr, nullptr);

    // flash attention (never materializes the 4096x4096 scores)
    flash_attn_kernel<<<B_ * (HW_ / 32), 256, 0, stream>>>(q_t, k_n, v_t, o_t);

    // output projection + bias + residual -> f32 d_out
    wmma_gemm_kernel<2><<<gg, 256, 0, stream>>>(wo_bf, o_t, bo, nullptr, x, out);
}
